// LightMutilHeadSelfAttention_8426725835335
// MI455X (gfx1250) — matrix operand order repaired and checked by reading, not compiled, not run on hardware
//
#include <hip/hip_runtime.h>

typedef _Float16 f16;
typedef __attribute__((ext_vector_type(16))) _Float16 v16h;
typedef __attribute__((ext_vector_type(8)))  _Float16 v8h;
typedef __attribute__((ext_vector_type(8)))  float    v8f;

union frag16 { v16h v; v8h p[2]; };

// A-matrix 16xK(=32) f16 fragment, row-major source, ISA 7.12.2 layout:
// lane<16: K=0..7 (halves0-7), K=16..23 (halves8-15); lane>=16: K=8..15, K=24..31
__device__ __forceinline__ v16h load_a_frag(const f16* rowptr, int lane) {
    const int kb = (lane >> 4) << 3;
    frag16 f;
    f.p[0] = *(const v8h*)(rowptr + kb);
    f.p[1] = *(const v8h*)(rowptr + 16 + kb);
    return f.v;
}
// B-matrix 32x16 f16 fragment from BT (N x K row-major) source:
// lane<16: K=0..7, K=16..23 of column n=lane; lane>=16: K=8..15, K=24..31
__device__ __forceinline__ v16h load_b_frag(const f16* colptr, int lane) {
    const int kb = (lane >> 4) << 3;
    frag16 f;
    f.p[0] = *(const v8h*)(colptr + kb);
    f.p[1] = *(const v8h*)(colptr + 16 + kb);
    return f.v;
}

// ---------------- pack / convert kernels ----------------

// x (B,C,N) fp32 -> Aq (B*N, C) f16
__global__ void pack_q_kernel(const float* __restrict__ x, f16* __restrict__ Aq) {
    int idx = blockIdx.x * 256 + threadIdx.x;          // < 4*256*3136
    int n = idx % 3136;
    int t = idx / 3136;                                 // b*256 + c
    int c = t & 255, b = t >> 8;
    Aq[((size_t)(b * 3136 + n)) * 256 + c] = (f16)x[idx];
}

// gather 2x2 stride-2 patches: Asr[b*784+p][i*4+ky*2+kx] = x[b,i,2py+ky,2px+kx]
__global__ void pack_sr_kernel(const float* __restrict__ x, f16* __restrict__ Asr) {
    int idx = blockIdx.x * 256 + threadIdx.x;          // < 4*784*1024
    int ci  = idx & 1023;
    int row = idx >> 10;                                // < 3136
    int b = row / 784, p = row - b * 784;
    int py = p / 28, px = p - py * 28;
    int i = ci >> 2, ky = (ci >> 1) & 1, kx = ci & 1;
    Asr[idx] = (f16)x[(((size_t)(b * 256 + i)) * 56 + (2 * py + ky)) * 56 + (2 * px + kx)];
}

// out (Nc x Kr) f16 = transpose of in (Kr x Nc) fp32
__global__ void convT_kernel(const float* __restrict__ in, f16* __restrict__ out, int Kr, int Nc) {
    int idx = blockIdx.x * 256 + threadIdx.x;          // < Kr*Nc
    int n = idx / Kr, k = idx - n * Kr;
    out[idx] = (f16)in[(size_t)k * Nc + n];
}

__global__ void convcp_kernel(const float* __restrict__ in, f16* __restrict__ out, int n) {
    int i = blockIdx.x * 256 + threadIdx.x;
    if (i < n) out[i] = (f16)in[i];
}

__global__ void zero16_kernel(f16* __restrict__ p, int n) {
    int i = blockIdx.x * 256 + threadIdx.x;
    if (i < n) p[i] = (f16)0.f;
}

// ---------------- generic WMMA GEMM ----------------
// C(M,N) = A(M,K) f16 row-major  @  BT(N,K) f16 row-major (i.e. B column-major)
// 128 threads = 4 waves, each wave one 16x16 tile. grid = (N/64, M/16).
// mode 0: f16 store, *alpha          mode 1: f32 store + bias
// mode 2: kv scatter -> kT/vT        mode 3: f32 + bias, scatter to (B,C,N)
__global__ void gemm16_kernel(const f16* __restrict__ A, const f16* __restrict__ BT,
                              const float* __restrict__ bias,
                              void* __restrict__ out, void* __restrict__ out2,
                              int M, int N, int K, int mode, float alpha) {
    const int wave = threadIdx.x >> 5, lane = threadIdx.x & 31;
    const int lan15 = lane & 15;
    const int colbase = (blockIdx.x * 4 + wave) * 16;
    const int rowbase = blockIdx.y * 16;
    const f16* ap = A  + (size_t)(rowbase + lan15) * K;
    const f16* bp = BT + (size_t)(colbase + lan15) * K;

    v8f acc = {};
    for (int k0 = 0; k0 < K; k0 += 32) {
        v16h a = load_a_frag(ap + k0, lane);
        v16h b = load_b_frag(bp + k0, lane);
        acc = __builtin_amdgcn_wmma_f32_16x16x32_f16(false, a, false, b,
                                                     (short)0, acc, false, false);
    }

    const int col   = colbase + lan15;
    const int rbase = rowbase + ((lane >> 4) << 3); // +8 rows for high half-wave

    if (mode == 0) {
        f16* o = (f16*)out;
        #pragma unroll
        for (int r = 0; r < 8; ++r)
            o[(size_t)(rbase + r) * N + col] = (f16)(acc[r] * alpha);
    } else if (mode == 1) {
        float* o = (float*)out;
        float bv = bias[col];
        #pragma unroll
        for (int r = 0; r < 8; ++r)
            o[(size_t)(rbase + r) * N + col] = acc[r] + bv;
    } else if (mode == 2) {
        // col = (d*8 + head)*2 + s ; k -> kT(B,8,800,32), v -> vT(B,8,32,800)
        f16* kO = (f16*)out;
        f16* vO = (f16*)out2;
        int s = col & 1, hh = (col >> 1) & 7, d = col >> 4;
        #pragma unroll
        for (int r = 0; r < 8; ++r) {
            int row = rbase + r;
            int bb = row / 784, m = row - bb * 784;
            f16 val = (f16)acc[r];
            if (s == 0) kO[((size_t)(bb * 8 + hh) * 800 + m) * 32 + d] = val;
            else        vO[((size_t)(bb * 8 + hh) * 32 + d) * 800 + m] = val;
        }
    } else { // mode 3: out(B,C,N) fp32 + bias
        float* o = (float*)out;
        float bv = bias[col];
        #pragma unroll
        for (int r = 0; r < 8; ++r) {
            int row = rbase + r;
            int bb = row / 3136, n = row - bb * 3136;
            o[((size_t)(bb * 256 + col)) * 3136 + n] = acc[r] + bv;
        }
    }
}

// ---------------- layernorm (row = 256 channels) ----------------
__global__ void lnorm_kernel(const float* __restrict__ xr, const float* __restrict__ g,
                             const float* __restrict__ bt, f16* __restrict__ xkv) {
    const int wave = threadIdx.x >> 5, lane = threadIdx.x & 31;
    const int row = blockIdx.x * 8 + wave;              // < 3136
    const float* p = xr + (size_t)row * 256;
    float v[8], s = 0.f, sq = 0.f;
    #pragma unroll
    for (int j = 0; j < 8; ++j) {
        v[j] = p[lane + 32 * j];
        s += v[j]; sq += v[j] * v[j];
    }
    #pragma unroll
    for (int o = 16; o >= 1; o >>= 1) {
        s  += __shfl_xor(s,  o, 32);
        sq += __shfl_xor(sq, o, 32);
    }
    float mean = s * (1.f / 256.f);
    float var  = sq * (1.f / 256.f) - mean * mean;
    float rstd = rsqrtf(var + 1e-5f);
    f16* o16 = xkv + (size_t)row * 256;
    #pragma unroll
    for (int j = 0; j < 8; ++j) {
        int c = lane + 32 * j;
        o16[c] = (f16)((v[j] - mean) * rstd * g[c] + bt[c]);
    }
}

// ---------------- fused attention ----------------
// block = (b, head, 32 query rows); 128 threads = 4 waves.
// QK^T (K=32, one wmma per 16x16 tile) + rel-pos bias -> LDS fp32,
// softmax over 784 cols, probs f16 -> LDS, PV (25 wmma K-steps / tile).
__global__ void attn_kernel(const f16* __restrict__ q, const f16* __restrict__ kT,
                            const f16* __restrict__ vT, const float* __restrict__ pos,
                            f16* __restrict__ outA) {
    __shared__ __align__(32) float scs[32 * 800];
    __shared__ __align__(32) f16  prb[32 * 800];

    const int n0   = blockIdx.x * 32;
    const int head = blockIdx.y;
    const int b    = blockIdx.z;
    const int bh   = b * 8 + head;
    const int wave = threadIdx.x >> 5, lane = threadIdx.x & 31;
    const int lan15 = lane & 15;
    const int hi8   = (lane >> 4) << 3;                 // 0 or 8 (C/D row offset)

    const f16* qbase = q  + (size_t)(b * 3136) * 256 + (size_t)head * 32;
    const f16* kbase = kT + (size_t)bh * 800 * 32;

    // ---- phase 1: scores = (q*scale) . k + bias ----
    for (int t = wave; t < 98; t += 4) {
        int rb = t / 49, cb = t - rb * 49;
        v16h a = load_a_frag(qbase + (size_t)(n0 + rb * 16 + lan15) * 256, lane);
        v16h bf = load_b_frag(kbase + (size_t)(cb * 16 + lan15) * 32, lane);
        v8f c = {};
        c = __builtin_amdgcn_wmma_f32_16x16x32_f16(false, a, false, bf,
                                                   (short)0, c, false, false);
        int m   = cb * 16 + lan15;
        int mdy = m / 56, mdx = m - mdy * 56;
        int rlo = rb * 16 + hi8;
        #pragma unroll
        for (int r = 0; r < 8; ++r) {
            int nr = n0 + rlo + r;
            int ny = nr / 56, nx = nr - ny * 56;
            float bias = pos[(mdy - ny + 55) * 111 + (mdx - nx + 55)];
            scs[(rlo + r) * 800 + m] = c[r] + bias;
        }
    }
    __syncthreads();

    // ---- phase 2: softmax over 784 cols (4 threads per row) ----
    {
        int r  = threadIdx.x >> 2;
        int p4 = threadIdx.x & 3;
        float mx = -1e30f;
        for (int m = p4; m < 784; m += 4) mx = fmaxf(mx, scs[r * 800 + m]);
        mx = fmaxf(mx, __shfl_xor(mx, 1, 32));
        mx = fmaxf(mx, __shfl_xor(mx, 2, 32));
        float s = 0.f;
        for (int m = p4; m < 784; m += 4) {
            float e = __expf(scs[r * 800 + m] - mx);
            scs[r * 800 + m] = e;
            s += e;
        }
        s += __shfl_xor(s, 1, 32);
        s += __shfl_xor(s, 2, 32);
        float inv = 1.f / s;
        for (int m = p4; m < 784; m += 4) prb[r * 800 + m] = (f16)(scs[r * 800 + m] * inv);
        for (int m = 784 + p4; m < 800; m += 4) prb[r * 800 + m] = (f16)0.f;
    }
    __syncthreads();

    // ---- phase 3: out = P @ V ----
    {
        int rb = wave >> 1, ct = wave & 1;
        const f16* pbase = &prb[(rb * 16 + lan15) * 800];
        const f16* vbase = vT + (size_t)bh * 32 * 800 + (size_t)(ct * 16 + lan15) * 800;
        v8f acc = {};
        for (int kk = 0; kk < 800; kk += 32) {
            v16h a  = load_a_frag(pbase + kk, lane);
            v16h bf = load_b_frag(vbase + kk, lane);
            acc = __builtin_amdgcn_wmma_f32_16x16x32_f16(false, a, false, bf,
                                                         (short)0, acc, false, false);
        }
        int col = head * 32 + ct * 16 + lan15;
        int rlo = rb * 16 + hi8;
        #pragma unroll
        for (int r = 0; r < 8; ++r) {
            int n = n0 + rlo + r;
            outA[((size_t)(b * 3136) + n) * 256 + col] = (f16)acc[r];
        }
    }
}

// ---------------- host launcher ----------------
extern "C" void kernel_launch(void* const* d_in, const int* in_sizes, int n_in,
                              void* d_out, int out_size, void* d_ws, size_t ws_size,
                              hipStream_t stream) {
    (void)in_sizes; (void)n_in; (void)out_size; (void)ws_size;
    const float* x      = (const float*)d_in[0];
    const float* Wq     = (const float*)d_in[1];
    const float* Wkv    = (const float*)d_in[2];
    const float* sr_w   = (const float*)d_in[3];
    const float* sr_b   = (const float*)d_in[4];
    const float* ln_g   = (const float*)d_in[5];
    const float* ln_b   = (const float*)d_in[6];
    const float* pos    = (const float*)d_in[7];
    const float* proj_w = (const float*)d_in[8];
    const float* proj_b = (const float*)d_in[9];
    float* out = (float*)d_out;

    char* ws = (char*)d_ws;
    size_t off = 0;
    auto carve = [&](size_t bytes) -> void* {
        void* p = ws + off;
        off += (bytes + 255) & ~(size_t)255;
        return p;
    };

    f16*   Aq   = (f16*)carve(3211264ull * 2);  // (B*N, 256)
    f16*   Asr  = (f16*)carve(3211264ull * 2);  // (B*784, 1024)
    f16*   WqT  = (f16*)carve(65536ull * 2);    // (256, 256)
    f16*   SrT  = (f16*)carve(262144ull * 2);   // (256, 1024)
    f16*   WkvT = (f16*)carve(131072ull * 2);   // (512, 256)
    f16*   PrT  = (f16*)carve(65536ull * 2);    // (256, 256)
    f16*   qbuf = (f16*)carve(3211264ull * 2);  // (B*N, 256), pre-scaled
    float* xr   = (float*)carve(802816ull * 4); // (B*784, 256)
    f16*   xkv  = (f16*)carve(802816ull * 2);   // (B*784, 256)
    f16*   kT   = (f16*)carve(819200ull * 2);   // (B,8,800,32)
    f16*   vT   = (f16*)carve(819200ull * 2);   // (B,8,32,800)
    f16*   aout = (f16*)carve(3211264ull * 2);  // (B*N, 256)

    const float scale = 0.17677669529663687f;   // 32^-0.5

    pack_q_kernel <<<12544, 256, 0, stream>>>(x, Aq);
    pack_sr_kernel<<<12544, 256, 0, stream>>>(x, Asr);
    convT_kernel  <<<256,   256, 0, stream>>>(Wq, WqT, 256, 256);
    convcp_kernel <<<1024,  256, 0, stream>>>(sr_w, SrT, 262144);
    convT_kernel  <<<512,   256, 0, stream>>>(Wkv, WkvT, 256, 512);
    convT_kernel  <<<256,   256, 0, stream>>>(proj_w, PrT, 256, 256);
    zero16_kernel <<<3200,  256, 0, stream>>>(kT, 819200);
    zero16_kernel <<<3200,  256, 0, stream>>>(vT, 819200);

    // q = (x_q @ Wq) * scale          -> f16
    gemm16_kernel<<<dim3(4, 784), 128, 0, stream>>>(Aq, WqT, nullptr, qbuf, nullptr,
                                                    12544, 256, 256, 0, scale);
    // xr = patches @ sr_w + sr_b      -> f32
    gemm16_kernel<<<dim3(4, 196), 128, 0, stream>>>(Asr, SrT, sr_b, xr, nullptr,
                                                    3136, 256, 1024, 1, 1.f);
    lnorm_kernel<<<392, 256, 0, stream>>>(xr, ln_g, ln_b, xkv);
    // kv = xkv @ Wkv -> scatter k/v
    gemm16_kernel<<<dim3(8, 196), 128, 0, stream>>>(xkv, WkvT, nullptr, kT, vT,
                                                    3136, 512, 256, 2, 1.f);
    attn_kernel<<<dim3(98, 8, 4), 128, 0, stream>>>(qbuf, kT, vT, pos, aout);
    // out = attn_out @ proj_w + proj_b -> (B,C,H,W) f32
    gemm16_kernel<<<dim3(4, 784), 128, 0, stream>>>(aout, PrT, proj_b, out, nullptr,
                                                    12544, 256, 256, 3, 1.f);
}
